// MbMlmLossLayer_11458972745999
// MI455X (gfx1250) — matrix operand order repaired and checked by reading, not compiled, not run on hardware
//
#include <hip/hip_runtime.h>
#include <hip/hip_bf16.h>
#include <math.h>

#define HID   256
#define VOC   50257
#define NTOK  4096
#define NTCOL 3142          // ceil(VOC/16)
#define LN_EPS 1e-3f
#define NEG_INF (-1e30f)

typedef __attribute__((ext_vector_type(16))) __bf16 v16bf;
typedef __attribute__((ext_vector_type(8)))  __bf16 v8bf;
typedef __attribute__((ext_vector_type(8)))  float  v8f;

// ---- f32 -> bf16 (round-to-nearest-even); used only in one-shot packing ----
__device__ __forceinline__ __bf16 f2bf(float f) {
  unsigned u = __builtin_bit_cast(unsigned, f);
  unsigned r = (u + 0x7FFFu + ((u >> 16) & 1u)) >> 16;
  unsigned short h = (unsigned short)r;
  return __builtin_bit_cast(__bf16, h);
}

// ---- WMMA A operand: 16x256 tile from bf16 eoutB, 8 x v16bf ---------------
// 16-bit A 16x32 layout: lane m (0-15): K {0..7,16..23}; lane m+16: K {8..15,24..31}
__device__ __forceinline__ void load_A16(v16bf* areg, const __bf16* __restrict__ eoutB,
                                         int r0, int lane) {
  const __bf16* rowp = eoutB + (size_t)(r0 + (lane & 15)) * HID;
  const int o = (lane < 16) ? 0 : 8;
#pragma unroll
  for (int kc = 0; kc < 8; ++kc) {
    v8bf a0 = *(const v8bf*)(rowp + kc * 32 + o);        // 16B aligned
    v8bf a1 = *(const v8bf*)(rowp + kc * 32 + 16 + o);
    areg[kc] = __builtin_shufflevector(a0, a1, 0, 1, 2, 3, 4, 5, 6, 7,
                                       8, 9, 10, 11, 12, 13, 14, 15);
  }
}

// ---- load all 8 B chunks of one 16-col tile (16 x global_load_b128) -------
// 16-bit B 32x16 layout: lane n (0-15): col n, K {0..7,16..23}; lane n+16: col n, K {8..15,24..31}
__device__ __forceinline__ void load_B8(v16bf* breg, const __bf16* __restrict__ embB,
                                        int cb, int lane) {
  int col  = cb + (lane & 15);
  int ccol = (col < VOC) ? col : (VOC - 1);   // clamp; results masked later
  const __bf16* bp = embB + (size_t)ccol * HID + ((lane < 16) ? 0 : 8);
#pragma unroll
  for (int kc = 0; kc < 8; ++kc) {
    v8bf x0 = *(const v8bf*)(bp + kc * 32);
    v8bf x1 = *(const v8bf*)(bp + kc * 32 + 16);
    breg[kc] = __builtin_shufflevector(x0, x1, 0, 1, 2, 3, 4, 5, 6, 7,
                                       8, 9, 10, 11, 12, 13, 14, 15);
  }
}

// ---- 8 chained WMMAs: full K=256 for one 16x16 tile -----------------------
__device__ __forceinline__ v8f wmma8(const v16bf* areg, const v16bf* breg) {
  v8f c = {0.f, 0.f, 0.f, 0.f, 0.f, 0.f, 0.f, 0.f};
#pragma unroll
  for (int kc = 0; kc < 8; ++kc)
    c = __builtin_amdgcn_wmma_f32_16x16x32_bf16(false, areg[kc], false, breg[kc],
                                                (short)0, c, false, false);
  return c;
}

// ---- per-lane online softmax update for one tile --------------------------
__device__ __forceinline__ void upd_stats(const v8f& c, int cb, int lane,
                                          const float* __restrict__ vbias,
                                          float* m, float* s) {
  int col = cb + (lane & 15);
  if (col < VOC) {
    float bias = vbias[col];
#pragma unroll
    for (int r = 0; r < 8; ++r) {
      float l  = c[r] + bias;
      float mn = fmaxf(m[r], l);
      s[r] = s[r] * __expf(m[r] - mn) + __expf(l - mn);
      m[r] = mn;
    }
  }
}

// ========================================================================
// Kernel 0: one-shot f32 -> bf16 repack of emb (25.7MB, L2-resident after)
// ========================================================================
__global__ __launch_bounds__(256) void cvt_emb_kernel(const float* __restrict__ emb,
                                                      __bf16* __restrict__ embB) {
  size_t i = ((size_t)blockIdx.x * 256 + threadIdx.x) * 8;
  if (i >= (size_t)VOC * HID) return;
  const float4* p = (const float4*)(emb + i);
  float4 f0 = p[0], f1 = p[1];
  v8bf o;
  o[0] = f2bf(f0.x); o[1] = f2bf(f0.y); o[2] = f2bf(f0.z); o[3] = f2bf(f0.w);
  o[4] = f2bf(f1.x); o[5] = f2bf(f1.y); o[6] = f2bf(f1.z); o[7] = f2bf(f1.w);
  *(v8bf*)(embB + i) = o;
}

// ========================================================================
// Kernel 1: experts + gates + select + LayerNorm + residual.
// Emits e_out in f32 (for the loss dot) and bf16 (WMMA A operand).
// ========================================================================
__global__ __launch_bounds__(256) void experts_kernel(
    const float* __restrict__ x, const int* __restrict__ xb,
    const float* __restrict__ W_sh, const float* __restrict__ b_sh,
    const float* __restrict__ W_sp, const float* __restrict__ b_sp,
    const float* __restrict__ w_gates,
    const float* __restrict__ ln_g, const float* __restrict__ ln_b,
    float* __restrict__ eout, __bf16* __restrict__ eoutB) {
  __shared__ float xst[HID * 16];     // transposed: [d][tok] -> broadcast reads
  __shared__ float obuf[16 * HID];    // [tok][h]
  __shared__ float gl[16 * 4];
  __shared__ float stats[16 * 2];
  __shared__ int   t4s[16];
  const int tid  = threadIdx.x;
  const int base = blockIdx.x * 16;

#pragma unroll
  for (int i = 0; i < 16; ++i)
    xst[tid * 16 + i] = x[(size_t)(base + i) * HID + tid];
  if (tid < 16) t4s[tid] = xb[base + tid] - 1;   // -1 => zero output
  __syncthreads();

  if (tid < 64) {                      // gate logits: (tok, e) pairs
    int tok = tid >> 2, e = tid & 3;
    int t4 = t4s[tok];
    float g = 0.f;
    if (t4 >= 0) {
      const float* wg = w_gates + (size_t)t4 * HID * 4 + e;   // [t][d][e]
      for (int d = 0; d < HID; ++d) g += xst[d * 16 + tok] * wg[d * 4];
    }
    gl[tok * 4 + e] = g;
  }
  __syncthreads();
  if (tid < 16) {                      // softmax over 4 experts
    float g0 = gl[tid*4], g1 = gl[tid*4+1], g2 = gl[tid*4+2], g3 = gl[tid*4+3];
    float mx = fmaxf(fmaxf(g0, g1), fmaxf(g2, g3));
    float e0 = __expf(g0-mx), e1 = __expf(g1-mx), e2 = __expf(g2-mx), e3 = __expf(g3-mx);
    float inv = 1.f / (e0 + e1 + e2 + e3);
    gl[tid*4] = e0*inv; gl[tid*4+1] = e1*inv; gl[tid*4+2] = e2*inv; gl[tid*4+3] = e3*inv;
  }
  __syncthreads();

  int t4r[16];
#pragma unroll
  for (int i = 0; i < 16; ++i) t4r[i] = t4s[i];

  float out[16];
#pragma unroll
  for (int i = 0; i < 16; ++i) out[i] = 0.f;

  for (int e = 0; e < 10; ++e) {
    const float* W = (e < 2) ? (W_sh + (size_t)e * HID * HID)
                             : (W_sp + (size_t)(e - 2) * HID * HID);
    float bv = (e < 2) ? b_sh[e * HID + tid] : b_sp[(e - 2) * HID + tid];
    float acc[16];
#pragma unroll
    for (int i = 0; i < 16; ++i) acc[i] = bv;
    for (int d = 0; d < HID; d += 4) {
#pragma unroll
      for (int dd = 0; dd < 4; ++dd) {
        float w = W[(size_t)(d + dd) * HID + tid];  // coalesced; L2 hit
        const float* xr = &xst[(d + dd) * 16];      // broadcast LDS reads
#pragma unroll
        for (int i = 0; i < 16; ++i) acc[i] += xr[i] * w;
      }
    }
    if (e < 2) {                                    // shared experts: gate e
#pragma unroll
      for (int i = 0; i < 16; ++i)
        if (t4r[i] >= 0) out[i] += gl[i * 4 + e] * acc[i];
    } else {                                        // branch experts 2t, 2t+1
      int spi = e - 2, t = spi >> 1, gix = 2 + (spi & 1);
#pragma unroll
      for (int i = 0; i < 16; ++i)
        if (t4r[i] == t) out[i] += gl[i * 4 + gix] * acc[i];
    }
  }

#pragma unroll
  for (int i = 0; i < 16; ++i) obuf[i * HID + tid] = out[i];
  __syncthreads();

  { // LayerNorm stats: 16 threads per token, shuffle-reduce inside half-wave
    int g = tid >> 4, l = tid & 15;
    float s = 0.f, s2 = 0.f;
#pragma unroll
    for (int j = 0; j < 16; ++j) {
      float v = obuf[g * HID + l + j * 16];
      s += v; s2 += v * v;
    }
#pragma unroll
    for (int mk = 1; mk <= 8; mk <<= 1) {
      s  += __shfl_xor(s, mk, 32);
      s2 += __shfl_xor(s2, mk, 32);
    }
    if (l == 0) {
      float mu = s / (float)HID;
      stats[g * 2]     = mu;
      stats[g * 2 + 1] = s2 / (float)HID - mu * mu;
    }
  }
  __syncthreads();

  float gv = ln_g[tid], bvv = ln_b[tid];
#pragma unroll
  for (int i = 0; i < 16; ++i) {
    float mu = stats[i * 2], var = stats[i * 2 + 1];
    float v  = (obuf[i * HID + tid] - mu) * rsqrtf(var + LN_EPS);
    float eo = gv * v + bvv + xst[tid * 16 + i];
    eout [(size_t)(base + i) * HID + tid] = eo;
    eoutB[(size_t)(base + i) * HID + tid] = f2bf(eo);
  }
}

// ========================================================================
// Kernel 2: online-softmax row stats (WMMA bf16), 8 waves/block, B tiles
// double-buffered: the 16 b128 loads of tile t+8 overlap the 8 WMMAs of
// tile t (tile guards are wave-uniform -> EXEC stays all-ones for WMMA).
// ========================================================================
__global__ __launch_bounds__(256) void rowstats_kernel(
    const __bf16* __restrict__ eoutB, const __bf16* __restrict__ embB,
    const float* __restrict__ vbias,
    float* __restrict__ row_m, float* __restrict__ row_s) {
  const int lane = threadIdx.x & 31;
  const int wave = threadIdx.x >> 5;
  const int r0   = blockIdx.x * 16;

  v16bf areg[8];
  load_A16(areg, eoutB, r0, lane);

  float m[8], s[8];
#pragma unroll
  for (int r = 0; r < 8; ++r) { m[r] = NEG_INF; s[r] = 0.f; }

  v16bf b0[8], b1[8];
  int ct = wave;                        // stride 8 waves, pipelined in pairs
  load_B8(b0, embB, ct * 16, lane);
  for (; ct < NTCOL; ct += 16) {
    load_B8(b1, embB, (ct + 8) * 16, lane);   // prefetch next tile (clamped)
    {
      v8f c = wmma8(areg, b0);
      upd_stats(c, ct * 16, lane, vbias, m, s);
    }
    if (ct + 8 < NTCOL) {
      load_B8(b0, embB, (ct + 16) * 16, lane);
      v8f c = wmma8(areg, b1);
      upd_stats(c, (ct + 8) * 16, lane, vbias, m, s);
    }
  }

  // merge (m,s) across the 16 lanes of each half-wave (masks stay in-half)
#pragma unroll
  for (int r = 0; r < 8; ++r) {
    float mr = m[r], sr = s[r];
#pragma unroll
    for (int mk = 1; mk <= 8; mk <<= 1) {
      float om = __shfl_xor(mr, mk, 32);
      float os = __shfl_xor(sr, mk, 32);
      float M  = fmaxf(mr, om);
      sr = sr * __expf(mr - M) + os * __expf(om - M);
      mr = M;
    }
    m[r] = mr; s[r] = sr;
  }

  __shared__ float s_m[8][16];
  __shared__ float s_s[8][16];
  if (lane == 0 || lane == 16) {
    int hb = (lane >> 4) * 8;     // lanes 0-15 own rows 0-7; 16-31 rows 8-15
#pragma unroll
    for (int r = 0; r < 8; ++r) { s_m[wave][hb + r] = m[r]; s_s[wave][hb + r] = s[r]; }
  }
  __syncthreads();
  if (threadIdx.x < 16) {
    float M = NEG_INF;
#pragma unroll
    for (int w = 0; w < 8; ++w) M = fmaxf(M, s_m[w][threadIdx.x]);
    float S = 0.f;
#pragma unroll
    for (int w = 0; w < 8; ++w) S += s_s[w][threadIdx.x] * __expf(s_m[w][threadIdx.x] - M);
    row_m[r0 + threadIdx.x] = M;
    row_s[r0 + threadIdx.x] = S;
  }
}

// ========================================================================
// Kernel 3: recompute logits (embB in L2) and stream prob to d_out,
// same double-buffered pipeline.
// ========================================================================
__global__ __launch_bounds__(256) void prob_kernel(
    const __bf16* __restrict__ eoutB, const __bf16* __restrict__ embB,
    const float* __restrict__ vbias,
    const float* __restrict__ row_m, const float* __restrict__ row_s,
    float* __restrict__ prob) {
  const int lane = threadIdx.x & 31;
  const int wave = threadIdx.x >> 5;
  const int r0   = blockIdx.x * 16;
  const int hi8  = (lane < 16) ? 0 : 8;   // C/D layout: VGPR r -> row r or r+8

  v16bf areg[8];
  load_A16(areg, eoutB, r0, lane);

  float rm[8], rsi[8];
#pragma unroll
  for (int r = 0; r < 8; ++r) {
    rm[r]  = row_m[r0 + hi8 + r];
    rsi[r] = 1.f / row_s[r0 + hi8 + r];
  }

  const int S = gridDim.y * 8;          // column-tile stride across all waves
  v16bf b0[8], b1[8];
  int ct = blockIdx.y * 8 + wave;
  load_B8(b0, embB, ct * 16, lane);
  for (; ct < NTCOL; ct += 2 * S) {
    load_B8(b1, embB, (ct + S) * 16, lane);
    {
      v8f c = wmma8(areg, b0);
      int col = ct * 16 + (lane & 15);
      if (col < VOC) {
        float bias = vbias[col];
#pragma unroll
        for (int r = 0; r < 8; ++r)
          prob[(size_t)(r0 + hi8 + r) * VOC + col] = __expf(c[r] + bias - rm[r]) * rsi[r];
      }
    }
    if (ct + S < NTCOL) {
      load_B8(b0, embB, (ct + 2 * S) * 16, lane);
      v8f c = wmma8(areg, b1);
      int col = (ct + S) * 16 + (lane & 15);
      if (col < VOC) {
        float bias = vbias[col];
#pragma unroll
        for (int r = 0; r < 8; ++r)
          prob[(size_t)(r0 + hi8 + r) * VOC + col] = __expf(c[r] + bias - rm[r]) * rsi[r];
      }
    }
  }
}

// ========================================================================
// Kernel 4: loss[n] = m + log(s) - (e_out[n] . emb[label] + bias[label])
// ========================================================================
__global__ __launch_bounds__(256) void loss_kernel(
    const float* __restrict__ eout, const float* __restrict__ emb,
    const float* __restrict__ vbias, const int* __restrict__ labels,
    const float* __restrict__ row_m, const float* __restrict__ row_s,
    float* __restrict__ loss) {
  int lane = threadIdx.x & 31, wave = threadIdx.x >> 5;
  int n   = blockIdx.x * 8 + wave;
  int lab = labels[n];
  const float* er = eout + (size_t)n   * HID;
  const float* em = emb  + (size_t)lab * HID;
  float acc = 0.f;
#pragma unroll
  for (int j = 0; j < 8; ++j) acc += er[lane * 8 + j] * em[lane * 8 + j];
#pragma unroll
  for (int mk = 1; mk <= 16; mk <<= 1) acc += __shfl_xor(acc, mk, 32);
  if (lane == 0)
    loss[n] = row_m[n] + logf(row_s[n]) - (acc + vbias[lab]);
}

// ========================================================================
extern "C" void kernel_launch(void* const* d_in, const int* in_sizes, int n_in,
                              void* d_out, int out_size, void* d_ws, size_t ws_size,
                              hipStream_t stream) {
  (void)in_sizes; (void)n_in; (void)out_size; (void)ws_size;
  const float* x       = (const float*)d_in[0];
  const int*   labels  = (const int*)  d_in[1];
  const int*   xb      = (const int*)  d_in[2];
  const float* W_sh    = (const float*)d_in[3];
  const float* b_sh    = (const float*)d_in[4];
  const float* W_sp    = (const float*)d_in[5];
  const float* b_sp    = (const float*)d_in[6];
  const float* w_gates = (const float*)d_in[7];
  const float* vbias   = (const float*)d_in[8];
  const float* emb     = (const float*)d_in[9];
  const float* ln_g    = (const float*)d_in[10];
  const float* ln_b    = (const float*)d_in[11];

  float* prob = (float*)d_out;
  float* loss = prob + (size_t)NTOK * VOC;

  // workspace layout (all 512B-aligned offsets)
  char* ws = (char*)d_ws;
  float*  eout  = (float*)ws;                                   // 4 MB
  float*  row_m = (float*)(ws + (size_t)NTOK * HID * 4);        // 16 KB
  float*  row_s = row_m + NTOK;                                 // 16 KB
  __bf16* eoutB = (__bf16*)(ws + (size_t)NTOK * HID * 4 + 2 * NTOK * 4);   // 2 MB
  __bf16* embB  = eoutB + (size_t)NTOK * HID;                   // 25.7 MB

  const size_t nemb  = (size_t)VOC * HID;
  const int    cvblk = (int)((nemb / 8 + 255) / 256);

  cvt_emb_kernel <<<cvblk, 256, 0, stream>>>(emb, embB);
  experts_kernel <<<NTOK / 16, 256, 0, stream>>>(x, xb, W_sh, b_sh, W_sp, b_sp,
                                                 w_gates, ln_g, ln_b, eout, eoutB);
  rowstats_kernel<<<NTOK / 16, 256, 0, stream>>>(eoutB, embB, vbias, row_m, row_s);
  prob_kernel    <<<dim3(NTOK / 16, 64), 256, 0, stream>>>(eoutB, embB, vbias,
                                                           row_m, row_s, prob);
  loss_kernel    <<<NTOK / 8, 256, 0, stream>>>(eout, emb, vbias, labels,
                                                row_m, row_s, loss);
}
